// MoE_26645977105052
// MI455X (gfx1250) — matrix operand order repaired and checked by reading, not compiled, not run on hardware
//
#include <hip/hip_runtime.h>
#include <hip/hip_bf16.h>
#include <math.h>

// ---------------- Problem constants ----------------
#define H_DIM 1024
#define F_DIM 4096
#define NEXP  8
#define TOPK  2

// GEMM tiling: block tile 128x128, K-step 32, 256 threads = 8 wave32.
// Wave tile 32x64 = 2 x 4 tiles of 16x16. WMMA: v_wmma_f32_16x16x32_bf16.
#define TILE_M 128
#define TILE_N 128
#define TILE_K 32
#define LDS_STRIDE 40   // 32 + 8 pad (bf16 elems)

typedef __bf16 bf16_t;
typedef bf16_t v16bf __attribute__((ext_vector_type(16)));
typedef bf16_t v8bf  __attribute__((ext_vector_type(8)));
typedef bf16_t v4bf  __attribute__((ext_vector_type(4)));
typedef float  v8f   __attribute__((ext_vector_type(8)));
typedef float  v4f   __attribute__((ext_vector_type(4)));

union Frag16 { v16bf v; v8bf h[2]; };

// ---------------- small utility kernels ----------------

__global__ void moe_zero_cnt(int* cnt) {
    if (threadIdx.x < NEXP) cnt[threadIdx.x] = 0;
}

__global__ void moe_xcast(const float* __restrict__ x, bf16_t* __restrict__ xb, int n4) {
    int i = blockIdx.x * blockDim.x + threadIdx.x;
    if (i >= n4) return;
    float4 v = ((const float4*)x)[i];
    union { bf16_t b[4]; uint2 u; } pk;
    pk.b[0] = (bf16_t)v.x; pk.b[1] = (bf16_t)v.y;
    pk.b[2] = (bf16_t)v.z; pk.b[3] = (bf16_t)v.w;
    ((uint2*)xb)[i] = pk.u;
}

// ---------------- Router ----------------
__global__ __launch_bounds__(256) void moe_router(
    const float* __restrict__ x, const float* __restrict__ gate_w,
    int* __restrict__ cnt, int* __restrict__ lists, float* __restrict__ wts,
    float* __restrict__ bpProbs, float* __restrict__ bpZ, int T)
{
    const int tid  = threadIdx.x;
    const int wave = tid >> 5;
    const int lane = tid & 31;
    const int t    = blockIdx.x * 8 + wave;

    __shared__ float shP[8][NEXP];
    __shared__ float shZ[8];

    float p[NEXP];
#pragma unroll
    for (int e = 0; e < NEXP; ++e) p[e] = 0.f;

    const float* xr = x + (size_t)t * H_DIM;
    for (int h = lane; h < H_DIM; h += 32) {
        float xv = xr[h];
        const float4* gw4 = (const float4*)(gate_w + (size_t)h * NEXP);
        float4 g0 = gw4[0], g1 = gw4[1];
        p[0] += xv * g0.x; p[1] += xv * g0.y; p[2] += xv * g0.z; p[3] += xv * g0.w;
        p[4] += xv * g1.x; p[5] += xv * g1.y; p[6] += xv * g1.z; p[7] += xv * g1.w;
    }
#pragma unroll
    for (int off = 16; off > 0; off >>= 1) {
#pragma unroll
        for (int e = 0; e < NEXP; ++e) p[e] += __shfl_xor(p[e], off, 32);
    }

    if (lane == 0) {
        float mx = p[0];
#pragma unroll
        for (int e = 1; e < NEXP; ++e) mx = fmaxf(mx, p[e]);
        float s = 0.f;
        float pr[NEXP];
#pragma unroll
        for (int e = 0; e < NEXP; ++e) { pr[e] = __expf(p[e] - mx); s += pr[e]; }
        float inv = __builtin_amdgcn_rcpf(s);
#pragma unroll
        for (int e = 0; e < NEXP; ++e) pr[e] *= inv;

        // top-2 (first occurrence wins ties, like lax.top_k)
        int e0 = 0; float p0 = pr[0];
#pragma unroll
        for (int e = 1; e < NEXP; ++e) if (pr[e] > p0) { p0 = pr[e]; e0 = e; }
        int e1 = -1; float p1 = -1.f;
#pragma unroll
        for (int e = 0; e < NEXP; ++e) if (e != e0 && pr[e] > p1) { p1 = pr[e]; e1 = e; }
        float rs = __builtin_amdgcn_rcpf(p0 + p1);
        float w0 = p0 * rs, w1 = p1 * rs;

        int pos0 = atomicAdd(&cnt[e0], 1);
        lists[e0 * T + pos0] = t * 2 + 0;
        wts[t * 2 + 0] = w0;
        int pos1 = atomicAdd(&cnt[e1], 1);
        lists[e1 * T + pos1] = t * 2 + 1;
        wts[t * 2 + 1] = w1;

        float lse = mx + __logf(s);
        shZ[wave] = lse * lse;
#pragma unroll
        for (int e = 0; e < NEXP; ++e) shP[wave][e] = pr[e];
    }
    __syncthreads();
    if (tid < NEXP) {            // fixed-order per-block reduction
        float s = 0.f;
#pragma unroll
        for (int w = 0; w < 8; ++w) s += shP[w][tid];
        bpProbs[blockIdx.x * NEXP + tid] = s;
    }
    if (tid == 0) {
        float s = 0.f;
#pragma unroll
        for (int w = 0; w < 8; ++w) s += shZ[w];
        bpZ[blockIdx.x] = s;
    }
}

// ---------------- fragment helpers ----------------

// A fragment (16x32 bf16): lanes 0-15 -> row m, K {0..7, 16..23};
// lanes 16-31 -> row m, K {8..15, 24..31}. Row-major LDS tile [M][LDS_STRIDE].
__device__ inline Frag16 load_a_frag(const bf16_t* sA, int mbase, int lane) {
    int half = lane >> 4;
    int m = mbase + (lane & 15);
    Frag16 f;
    f.h[0] = *(const v8bf*)(sA + m * LDS_STRIDE + 8 * half);
    f.h[1] = *(const v8bf*)(sA + m * LDS_STRIDE + 16 + 8 * half);
    return f;
}

// B fragment (32x16 bf16): lane = n + 16*khalf holds column n, K {8*khalf..+7, 16+8*khalf..+7}.
// LDS tile stored transposed as [N][LDS_STRIDE] (i.e. W^T), contiguous loads.
__device__ inline Frag16 load_b_frag(const bf16_t* sB, int nbase, int lane) {
    int khalf = lane >> 4;
    int n = nbase + (lane & 15);
    Frag16 f;
    f.h[0] = *(const v8bf*)(sB + n * LDS_STRIDE + 8 * khalf);
    f.h[1] = *(const v8bf*)(sB + n * LDS_STRIDE + 16 + 8 * khalf);
    return f;
}

// fp32 weight tile [TILE_K x TILE_N] at (k0, n0) -> bf16 transposed LDS [n][k].
// Each thread owns a 4x4 block: 4 coalesced float4 row loads, register
// transpose (v_cvt_pk_bf16_f32), ds_store_b64 pairs. Also issues a
// speculative global_prefetch for the next K-step's tile to warm L2/WGP$
// (safe out-of-bounds: speculative prefetch drops translation failures).
__device__ inline void stage_b_tile(const float* __restrict__ w, int ldw,
                                    int k0, int n0, bf16_t* sB, int tid) {
    int nc = tid & 31;        // column group (4 cols) -> coalesced across wave
    int kq = tid >> 5;        // 4-row group (8 waves x 4 = 32 k rows)
    const float* base = w + (size_t)(k0 + kq * 4) * ldw + n0 + nc * 4;
    __builtin_prefetch(base + (size_t)TILE_K * ldw, 0, 0);   // next K-step
    float4 r0 = *(const float4*)(base);
    float4 r1 = *(const float4*)(base + ldw);
    float4 r2 = *(const float4*)(base + 2 * (size_t)ldw);
    float4 r3 = *(const float4*)(base + 3 * (size_t)ldw);
    const float* c0 = &r0.x; const float* c1 = &r1.x;
    const float* c2 = &r2.x; const float* c3 = &r3.x;
#pragma unroll
    for (int j = 0; j < 4; ++j) {
        v4bf v;
        v[0] = (bf16_t)c0[j]; v[1] = (bf16_t)c1[j];
        v[2] = (bf16_t)c2[j]; v[3] = (bf16_t)c3[j];
        *(v4bf*)(sB + (nc * 4 + j) * LDS_STRIDE + kq * 4) = v;
    }
}

// gathered activation rows (bf16, row stride ldx) -> LDS [row][k] using the
// CDNA5 direct-to-LDS async path (ASYNCcnt). INST_OFFSET is added to both the
// global and LDS addresses, so one base covers both 16B chunks.
__device__ inline void stage_a_tile_async(const bf16_t* __restrict__ act, int ldx,
                                          const int* sEnt, int k0, bf16_t* sA, int tid) {
    int row  = tid >> 1;
    int half = tid & 1;
    int ent  = sEnt[row];
    const bf16_t* src = act + (size_t)ent * ldx + k0 + half * 16;
    unsigned long long ga = (unsigned long long)(uintptr_t)src;
    unsigned ldsOff = (unsigned)(uintptr_t)(sA + row * LDS_STRIDE + half * 16);
    asm volatile(
        "global_load_async_to_lds_b128 %0, %1, off\n\t"
        "global_load_async_to_lds_b128 %0, %1, off offset:16"
        :: "v"(ldsOff), "v"(ga) : "memory");
}

__device__ inline void wait_async_lds() {
    asm volatile("s_wait_asynccnt 0x0" ::: "memory");
}

__device__ inline float fast_silu_mul(float g, float u) {
    return g * __builtin_amdgcn_rcpf(1.f + __expf(-g)) * u;
}

// ---------------- FFN stage 1: h = silu(x@Wg) * (x@Wu), bf16 out ----------------
__global__ __launch_bounds__(256) void moe_ffn1(
    const float* __restrict__ w_gate, const float* __restrict__ w_up,
    const bf16_t* __restrict__ xb, const int* __restrict__ lists,
    const int* __restrict__ cnt, bf16_t* __restrict__ hbuf, int T)
{
    const int e  = blockIdx.z;
    const int m0 = blockIdx.y * TILE_M;
    const int n0 = blockIdx.x * TILE_N;
    const int nE = cnt[e];
    if (m0 >= nE) return;

    const int tid  = threadIdx.x;
    const int wave = tid >> 5;
    const int lane = tid & 31;
    const int waveM = (wave & 3) * 32;   // 4 waves along M
    const int waveN = (wave >> 2) * 64;  // 2 waves along N

    __shared__ __align__(16) bf16_t sA[TILE_M * LDS_STRIDE];
    __shared__ __align__(16) bf16_t sBg[TILE_N * LDS_STRIDE];
    __shared__ __align__(16) bf16_t sBu[TILE_N * LDS_STRIDE];
    __shared__ int sEntTok[TILE_M];   // token id per tile row
    __shared__ int sEntRow[TILE_M];   // packed entry id (t*2+slot)

    if (tid < TILE_M) {
        int gm  = m0 + tid;
        int ent = lists[(size_t)e * T + (gm < nE ? gm : nE - 1)];
        sEntRow[tid] = ent;
        sEntTok[tid] = ent >> 1;
    }

    const float* wgE = w_gate + (size_t)e * H_DIM * F_DIM;
    const float* wuE = w_up   + (size_t)e * H_DIM * F_DIM;

    v8f accG[2][4], accU[2][4];
#pragma unroll
    for (int mf = 0; mf < 2; ++mf)
#pragma unroll
        for (int nf = 0; nf < 4; ++nf) { accG[mf][nf] = (v8f)0.f; accU[mf][nf] = (v8f)0.f; }

    for (int k0 = 0; k0 < H_DIM; k0 += TILE_K) {
        __syncthreads();
        stage_a_tile_async(xb, H_DIM, sEntTok, k0, sA, tid);
        stage_b_tile(wgE, F_DIM, k0, n0, sBg, tid);
        stage_b_tile(wuE, F_DIM, k0, n0, sBu, tid);
        wait_async_lds();
        __syncthreads();

        Frag16 fa[2];
#pragma unroll
        for (int mf = 0; mf < 2; ++mf) fa[mf] = load_a_frag(sA, waveM + mf * 16, lane);
#pragma unroll
        for (int nf = 0; nf < 4; ++nf) {
            Frag16 bg = load_b_frag(sBg, waveN + nf * 16, lane);
            Frag16 bu = load_b_frag(sBu, waveN + nf * 16, lane);
#pragma unroll
            for (int mf = 0; mf < 2; ++mf) {
                accG[mf][nf] = __builtin_amdgcn_wmma_f32_16x16x32_bf16(
                    false, fa[mf].v, false, bg.v, (short)0, accG[mf][nf], false, false);
                accU[mf][nf] = __builtin_amdgcn_wmma_f32_16x16x32_bf16(
                    false, fa[mf].v, false, bu.v, (short)0, accU[mf][nf], false, false);
            }
        }
    }

    // epilogue: h = silu(g) * u, scatter rows by packed entry id
    const int half = lane >> 4;
    const int ncol = lane & 15;
#pragma unroll
    for (int mf = 0; mf < 2; ++mf) {
#pragma unroll
        for (int nf = 0; nf < 4; ++nf) {
            v8f g = accG[mf][nf], u = accU[mf][nf];
#pragma unroll
            for (int i = 0; i < 8; ++i) {
                int m  = waveM + mf * 16 + half * 8 + i;
                int gm = m0 + m;
                if (gm < nE) {
                    float hv = fast_silu_mul(g[i], u[i]);
                    int n = n0 + waveN + nf * 16 + ncol;
                    hbuf[(size_t)sEntRow[m] * F_DIM + n] = (bf16_t)hv;
                }
            }
        }
    }
}

// ---------------- FFN stage 2: contrib[slot] = w_slot * (h @ Wd) ----------------
__global__ __launch_bounds__(256) void moe_ffn2(
    const float* __restrict__ w_down, const bf16_t* __restrict__ hbuf,
    const int* __restrict__ lists, const int* __restrict__ cnt,
    const float* __restrict__ wts, float* __restrict__ contrib, int T)
{
    const int e  = blockIdx.z;
    const int m0 = blockIdx.y * TILE_M;
    const int n0 = blockIdx.x * TILE_N;
    const int nE = cnt[e];
    if (m0 >= nE) return;

    const int tid  = threadIdx.x;
    const int wave = tid >> 5;
    const int lane = tid & 31;
    const int waveM = (wave & 3) * 32;
    const int waveN = (wave >> 2) * 64;

    __shared__ __align__(16) bf16_t sA[TILE_M * LDS_STRIDE];
    __shared__ __align__(16) bf16_t sBd[TILE_N * LDS_STRIDE];
    __shared__ int sEntRow[TILE_M];

    if (tid < TILE_M) {
        int gm = m0 + tid;
        sEntRow[tid] = lists[(size_t)e * T + (gm < nE ? gm : nE - 1)];
    }

    const float* wdE = w_down + (size_t)e * F_DIM * H_DIM;

    v8f acc[2][4];
#pragma unroll
    for (int mf = 0; mf < 2; ++mf)
#pragma unroll
        for (int nf = 0; nf < 4; ++nf) acc[mf][nf] = (v8f)0.f;

    for (int k0 = 0; k0 < F_DIM; k0 += TILE_K) {
        __syncthreads();
        stage_a_tile_async(hbuf, F_DIM, sEntRow, k0, sA, tid);
        stage_b_tile(wdE, H_DIM, k0, n0, sBd, tid);
        wait_async_lds();
        __syncthreads();

        Frag16 fa[2];
#pragma unroll
        for (int mf = 0; mf < 2; ++mf) fa[mf] = load_a_frag(sA, waveM + mf * 16, lane);
#pragma unroll
        for (int nf = 0; nf < 4; ++nf) {
            Frag16 bd = load_b_frag(sBd, waveN + nf * 16, lane);
#pragma unroll
            for (int mf = 0; mf < 2; ++mf) {
                acc[mf][nf] = __builtin_amdgcn_wmma_f32_16x16x32_bf16(
                    false, fa[mf].v, false, bd.v, (short)0, acc[mf][nf], false, false);
            }
        }
    }

    const int half = lane >> 4;
    const int ncol = lane & 15;
#pragma unroll
    for (int mf = 0; mf < 2; ++mf) {
#pragma unroll
        for (int nf = 0; nf < 4; ++nf) {
            v8f a = acc[mf][nf];
#pragma unroll
            for (int i = 0; i < 8; ++i) {
                int m  = waveM + mf * 16 + half * 8 + i;
                int gm = m0 + m;
                if (gm < nE) {
                    int ent = sEntRow[m];
                    float w = wts[ent];
                    int n = n0 + waveN + nf * 16 + ncol;
                    contrib[(size_t)ent * H_DIM + n] = a[i] * w;
                }
            }
        }
    }
}

// ---------------- combine: out[t] = contrib[2t] + contrib[2t+1] ----------------
__global__ void moe_combine(const float* __restrict__ contrib,
                            float* __restrict__ out, int total4) {
    int i = blockIdx.x * blockDim.x + threadIdx.x;
    if (i >= total4) return;
    const int h4pr = H_DIM / 4;
    int t  = i / h4pr;
    int h4 = i - t * h4pr;
    const v4f* c4 = (const v4f*)contrib;
    v4f a = c4[(size_t)(2 * t) * h4pr + h4];
    v4f b = c4[(size_t)(2 * t + 1) * h4pr + h4];
    v4f o = a + b;
    __builtin_nontemporal_store(o, (v4f*)out + i);   // out never re-read on device
}

// ---------------- finalize aux losses (fixed-order reductions) ----------------
__global__ void moe_finalize(const float* __restrict__ bpProbs,
                             const float* __restrict__ bpZ,
                             const int* __restrict__ cnt,
                             float* __restrict__ losses, int nBlocks, int T) {
    __shared__ float sp[NEXP];
    int tid = threadIdx.x;
    if (tid < NEXP) {
        float s = 0.f;
        for (int b = 0; b < nBlocks; ++b) s += bpProbs[b * NEXP + tid];
        sp[tid] = s;
    }
    __syncthreads();
    if (tid == 0) {
        float bal = 0.f;
        for (int e = 0; e < NEXP; ++e) {
            float frac  = (float)cnt[e] / (float)(T * TOPK);
            float meanp = sp[e] / (float)T;
            bal += frac * meanp;
        }
        losses[0] = (float)NEXP * bal;
        float z = 0.f;
        for (int b = 0; b < nBlocks; ++b) z += bpZ[b];
        losses[1] = z / (float)T;
    }
}

// ---------------- host launcher ----------------
extern "C" void kernel_launch(void* const* d_in, const int* in_sizes, int n_in,
                              void* d_out, int out_size, void* d_ws, size_t ws_size,
                              hipStream_t stream) {
    const float* x      = (const float*)d_in[0];
    const float* gate_w = (const float*)d_in[1];
    const float* w_gate = (const float*)d_in[2];
    const float* w_up   = (const float*)d_in[3];
    const float* w_down = (const float*)d_in[4];
    float* out = (float*)d_out;

    const int T = in_sizes[0] / H_DIM;           // 8192 tokens
    const int nRouterBlocks = T / 8;

    // workspace carve-out (256B aligned)
    size_t o = 0;
    auto take = [&](size_t bytes) { size_t r = o; o = (o + bytes + 255) & ~(size_t)255; return r; };
    char* ws = (char*)d_ws;
    int*    cnt     = (int*)   (ws + take(NEXP * sizeof(int)));
    float*  wts     = (float*) (ws + take((size_t)2 * T * sizeof(float)));
    int*    lists   = (int*)   (ws + take((size_t)NEXP * T * sizeof(int)));
    float*  bpP     = (float*) (ws + take((size_t)nRouterBlocks * NEXP * sizeof(float)));
    float*  bpZ     = (float*) (ws + take((size_t)nRouterBlocks * sizeof(float)));
    bf16_t* xb      = (bf16_t*)(ws + take((size_t)T * H_DIM * sizeof(bf16_t)));
    bf16_t* hbuf    = (bf16_t*)(ws + take((size_t)2 * T * F_DIM * sizeof(bf16_t)));
    float*  contrib = (float*) (ws + take((size_t)2 * T * H_DIM * sizeof(float)));
    (void)ws_size;

    moe_zero_cnt<<<1, 64, 0, stream>>>(cnt);

    moe_router<<<nRouterBlocks, 256, 0, stream>>>(x, gate_w, cnt, lists, wts, bpP, bpZ, T);

    int n4 = (T * H_DIM) / 4;
    moe_xcast<<<(n4 + 255) / 256, 256, 0, stream>>>(x, xb, n4);

    dim3 g1(F_DIM / TILE_N, T / TILE_M, NEXP);   // 32 x 64 x 8 (most exit on count guard)
    moe_ffn1<<<g1, 256, 0, stream>>>(w_gate, w_up, xb, lists, cnt, hbuf, T);

    dim3 g2(H_DIM / TILE_N, T / TILE_M, NEXP);   // 8 x 64 x 8
    moe_ffn2<<<g2, 256, 0, stream>>>(w_down, hbuf, lists, cnt, wts, contrib, T);

    moe_combine<<<(n4 + 255) / 256, 256, 0, stream>>>(contrib, out, n4);

    moe_finalize<<<1, 64, 0, stream>>>(bpP, bpZ, cnt, out + (size_t)T * H_DIM, nRouterBlocks, T);
}
